// GPT_5592047419606
// MI455X (gfx1250) — compile-verified, matrix operand order checked by reading, not run on hardware
//
#include <hip/hip_runtime.h>
#include <hip/hip_bf16.h>
#include <math.h>

typedef __bf16 bf16;
typedef bf16  v16bf __attribute__((ext_vector_type(16)));
typedef bf16  v8bf  __attribute__((ext_vector_type(8)));
typedef float v8f   __attribute__((ext_vector_type(8)));

// Problem constants
constexpr int CB   = 4;
constexpr int CTS  = 256;
constexpr int CD   = 1024;
constexpr int CH   = 16;
constexpr int CL   = 4;
constexpr int CV   = 32000;
constexpr int CM   = 3 * CTS - 1;   // 767 tokens
constexpr int CDH  = 64;
constexpr int CFF  = 4096;
constexpr int CROWS  = CB * CM;     // 3068
constexpr int CROWSP = 3072;        // padded to multiple of 256
constexpr int CHROWS = CB * CTS;    // 1024 gathered head rows

// LDS tile geometry for GEMM W staging
constexpr int TROW = 144;           // 64 k-elems * 2B = 128B, padded to 144B
constexpr int TBUF = 64 * TROW;     // one 64x64 bf16 tile (padded) = 9216B

// ---------------------------------------------------------------------------
// fp32 -> bf16 conversion (grid-stride)
// ---------------------------------------------------------------------------
__global__ void f2bf_kernel(const float* __restrict__ s, bf16* __restrict__ d, int n) {
    int i = blockIdx.x * blockDim.x + threadIdx.x;
    int stride = gridDim.x * blockDim.x;
    for (; i < n; i += stride) d[i] = (bf16)s[i];
}

// ---------------------------------------------------------------------------
// zero the padding rows of residual stream + attention output
// ---------------------------------------------------------------------------
__global__ void zero_pad_kernel(float* __restrict__ Xf, bf16* __restrict__ Yb) {
    int i = blockIdx.x * blockDim.x + threadIdx.x;
    const int n = (CROWSP - CROWS) * CD;
    if (i < n) {
        Xf[(size_t)CROWS * CD + i] = 0.0f;
        Yb[(size_t)CROWS * CD + i] = (bf16)0.0f;
    }
}

// ---------------------------------------------------------------------------
// token + position embedding: x[b,m,:] = tok + gpos[timesteps[b]] + pos_emb[m]
// ---------------------------------------------------------------------------
__global__ void embed_kernel(const float* __restrict__ states,
                             const int*   __restrict__ actions,
                             const float* __restrict__ rtg,
                             const int*   __restrict__ timesteps,
                             const float* __restrict__ pos_emb,
                             const float* __restrict__ gpos,
                             const float* __restrict__ Ws, const float* __restrict__ bs,
                             const float* __restrict__ Wr, const float* __restrict__ br,
                             const float* __restrict__ aemb,
                             float* __restrict__ X) {
    const int m = blockIdx.x;
    const int b = blockIdx.y;
    const int ts = timesteps[b];
    const int r = m % 3, i = m / 3;
    float* xr = X + ((size_t)b * CM + m) * CD;
    const float* gp = gpos + (size_t)ts * CD;
    const float* pp = pos_emb + (size_t)m * CD;
#pragma unroll
    for (int j = 0; j < 4; ++j) {
        int d = threadIdx.x + 256 * j;
        float val;
        if (r == 0)      val = tanhf(rtg[b * CTS + i] * Wr[d] + br[d]);
        else if (r == 1) val = tanhf(states[b * CTS + i] * Ws[d] + bs[d]);
        else {
            int a = actions[b * CTS + i + 1];
            val = tanhf(aemb[(size_t)a * CD + d]);
        }
        xr[d] = val + gp[d] + pp[d];
    }
}

// ---------------------------------------------------------------------------
// LayerNorm over D=1024, output bf16 (feeds WMMA GEMM A operand)
// ---------------------------------------------------------------------------
__global__ void ln_kernel(const float* __restrict__ X,
                          const float* __restrict__ g, const float* __restrict__ be,
                          bf16* __restrict__ Hb) {
    __shared__ float red[256];
    const int row = blockIdx.x;
    const int tid = threadIdx.x;
    const float* xr = X + (size_t)row * CD;
    float vv[4];
    float s = 0.f;
#pragma unroll
    for (int i = 0; i < 4; ++i) { vv[i] = xr[tid + 256 * i]; s += vv[i]; }
    red[tid] = s; __syncthreads();
    for (int o = 128; o > 0; o >>= 1) { if (tid < o) red[tid] += red[tid + o]; __syncthreads(); }
    const float mean = red[0] * (1.0f / CD);
    __syncthreads();
    float s2 = 0.f;
#pragma unroll
    for (int i = 0; i < 4; ++i) { float d = vv[i] - mean; s2 += d * d; }
    red[tid] = s2; __syncthreads();
    for (int o = 128; o > 0; o >>= 1) { if (tid < o) red[tid] += red[tid + o]; __syncthreads(); }
    const float rs = rsqrtf(red[0] * (1.0f / CD) + 1e-5f);
    bf16* hr = Hb + (size_t)row * CD;
#pragma unroll
    for (int i = 0; i < 4; ++i) {
        int d = tid + 256 * i;
        hr[d] = (bf16)((vv[i] - mean) * rs * g[d] + be[d]);
    }
}

// ---------------------------------------------------------------------------
// WMMA bf16 GEMM: out[m,n] = sum_k A[m,k]*W[n,k] (+bias) (+residual) (gelu?)
// A: [Mrows,K] bf16 streamed from global; W: [N,K] bf16, 64x64 tile staged in
// LDS via async global->LDS loads (double buffered), shared by all 8 waves.
// Block = 256 threads = 8 waves; wave tile = 32x64. All four B fragments of a
// 32-wide K sub-step are hoisted into registers before the 8-WMMA chain so
// LDS latency is paid once per sub-step, not once per n-tile.
// grid = (N/64, Mrows/256).
// ---------------------------------------------------------------------------
template <bool GELU>
__global__ __launch_bounds__(256)
void gemm_bf16_kernel(const bf16* __restrict__ A,
                      const bf16* __restrict__ W,
                      const float* __restrict__ bias,
                      const float* __restrict__ res,
                      float* __restrict__ outF,
                      bf16*  __restrict__ outB,
                      int N, int K) {
    __shared__ alignas(16) char smem[2 * TBUF];

    const int tid  = threadIdx.x;
    const int lane = tid & 31;
    const int wave = tid >> 5;
    const int half = lane >> 4;      // lane group 0..15 vs 16..31
    const int lrow = lane & 15;
    const int mbase = blockIdx.y * 256 + wave * 32;
    const int nbase = blockIdx.x * 64;

    // A fragment pointers: two 16-row tiles per wave
    const bf16* a0 = A + (size_t)(mbase +  0 + lrow) * K + 8 * half;
    const bf16* a1 = A + (size_t)(mbase + 16 + lrow) * K + 8 * half;

    // Async staging: 64 rows x 64 k = 512 16B-chunks; thread does chunks tid, tid+256.
    // chunk c: row = c>>3, kc = c&7 -> global (nbase+row)*K + kb + kc*8,
    //          lds  row*TROW + kc*16
    const int r0 = tid >> 3, kc = tid & 7;
    const int r1 = r0 + 32;
    const bf16* wg0 = W + (size_t)(nbase + r0) * K + kc * 8;
    const bf16* wg1 = W + (size_t)(nbase + r1) * K + kc * 8;
    // Low 32 bits of a generic LDS pointer are the LDS byte address (ISA 10.2)
    const unsigned lds0 = (unsigned)(size_t)(smem) + r0 * TROW + kc * 16;
    const unsigned lds1 = (unsigned)(size_t)(smem) + r1 * TROW + kc * 16;

    // prefetch first W tile into buffer 0
    asm volatile("global_load_async_to_lds_b128 %0, %1, off"
                 :: "v"(lds0), "v"(wg0) : "memory");
    asm volatile("global_load_async_to_lds_b128 %0, %1, off"
                 :: "v"(lds1), "v"(wg1) : "memory");

    v8f c[8] = {};   // c[2*t + mt]: n-tile t (0..3), m-tile mt (0..1)
    int buf = 0;

    for (int kb = 0; kb < K; kb += 64) {
        // wait for this wave's async copies, then make tile visible block-wide
        asm volatile("s_wait_asynccnt 0" ::: "memory");
        __syncthreads();
        const int cur = buf;
        if (kb + 64 < K) {
            const int nb = buf ^ 1;
            asm volatile("global_load_async_to_lds_b128 %0, %1, off"
                         :: "v"(lds0 + nb * TBUF), "v"(wg0 + kb + 64) : "memory");
            asm volatile("global_load_async_to_lds_b128 %0, %1, off"
                         :: "v"(lds1 + nb * TBUF), "v"(wg1 + kb + 64) : "memory");
        }

#pragma unroll
        for (int s = 0; s < 2; ++s) {           // two 32-wide K sub-steps
            const int ka = kb + 32 * s;
            __builtin_prefetch(a0 + ka + 128, 0, 1);
            v8bf a0lo = *(const v8bf*)(a0 + ka);
            v8bf a0hi = *(const v8bf*)(a0 + ka + 16);
            v16bf aF0 = __builtin_shufflevector(a0lo, a0hi,
                        0,1,2,3,4,5,6,7,8,9,10,11,12,13,14,15);
            v8bf a1lo = *(const v8bf*)(a1 + ka);
            v8bf a1hi = *(const v8bf*)(a1 + ka + 16);
            v16bf aF1 = __builtin_shufflevector(a1lo, a1hi,
                        0,1,2,3,4,5,6,7,8,9,10,11,12,13,14,15);

            // hoist all four B fragments into distinct registers first
            v16bf bF[4];
#pragma unroll
            for (int t = 0; t < 4; ++t) {
                const char* bp = smem + cur * TBUF + (16 * t + lrow) * TROW
                               + s * 64 + 16 * half;
                v8bf blo = *(const v8bf*)bp;
                v8bf bhi = *(const v8bf*)(bp + 32);
                bF[t] = __builtin_shufflevector(blo, bhi,
                        0,1,2,3,4,5,6,7,8,9,10,11,12,13,14,15);
            }
            // 8 WMMAs back-to-back
#pragma unroll
            for (int t = 0; t < 4; ++t) {
                c[2*t+0] = __builtin_amdgcn_wmma_f32_16x16x32_bf16(
                               false, aF0, false, bF[t], (short)0, c[2*t+0], false, false);
                c[2*t+1] = __builtin_amdgcn_wmma_f32_16x16x32_bf16(
                               false, aF1, false, bF[t], (short)0, c[2*t+1], false, false);
            }
        }
        buf ^= 1;
    }

    // Epilogue. C/D layout: VGPR r -> M = r + 8*half (per lane group), N = lrow.
#pragma unroll
    for (int t = 0; t < 4; ++t) {
        const int n = nbase + 16 * t + lrow;
        const float bv = bias ? bias[n] : 0.0f;
#pragma unroll
        for (int mt = 0; mt < 2; ++mt) {
#pragma unroll
            for (int r = 0; r < 8; ++r) {
                const int m = mbase + 16 * mt + 8 * half + r;
                const size_t idx = (size_t)m * N + n;
                float v = c[2*t+mt][r] + bv;
                if (res)  v += res[idx];
                if (GELU) v = 0.5f * v * (1.0f + erff(v * 0.70710678f));
                if (outF) outF[idx] = v;
                if (outB) outB[idx] = (bf16)v;
            }
        }
    }
}

// ---------------------------------------------------------------------------
// Causal attention, one wave per (b,h,q). DH=64: lane owns 2 dims for output,
// keys strided by lane for score/softmax; all cross-lane traffic via shuffles.
// ---------------------------------------------------------------------------
__global__ void attn_kernel(const float* __restrict__ Q,
                            const float* __restrict__ K,
                            const float* __restrict__ Vv,
                            bf16* __restrict__ Y) {
    const int wave = threadIdx.x >> 5;
    const int lane = threadIdx.x & 31;
    const int q = blockIdx.x * 4 + wave;
    if (q >= CM) return;
    const int b = blockIdx.z, h = blockIdx.y;
    const size_t brow = (size_t)b * CM;
    const int base = h * CDH;

    const float4* qp = (const float4*)(Q + (brow + q) * CD + base);
    float4 qv[16];
#pragma unroll
    for (int i = 0; i < 16; ++i) qv[i] = qp[i];

    float sreg[24];
    float mloc = -1e30f;
#pragma unroll
    for (int j = 0; j < 24; ++j) {
        const int k = j * 32 + lane;
        float sc = -1e30f;
        if (k <= q) {
            const float4* kp = (const float4*)(K + (brow + k) * CD + base);
            float acc = 0.f;
#pragma unroll
            for (int i = 0; i < 16; ++i) {
                float4 kk = kp[i];
                acc += qv[i].x * kk.x + qv[i].y * kk.y + qv[i].z * kk.z + qv[i].w * kk.w;
            }
            sc = acc * 0.125f;   // 1/sqrt(64)
        }
        sreg[j] = sc;
        mloc = fmaxf(mloc, sc);
    }
    for (int o = 16; o > 0; o >>= 1) mloc = fmaxf(mloc, __shfl_xor(mloc, o, 32));

    float ssum = 0.f;
#pragma unroll
    for (int j = 0; j < 24; ++j) {
        const int k = j * 32 + lane;
        const float e = (k <= q) ? __expf(sreg[j] - mloc) : 0.0f;
        sreg[j] = e;
        ssum += e;
    }
    for (int o = 16; o > 0; o >>= 1) ssum += __shfl_xor(ssum, o, 32);
    const float inv = 1.0f / ssum;

    const int d0 = base + 2 * lane;
    float acc0 = 0.f, acc1 = 0.f;
#pragma unroll
    for (int j = 0; j < 24; ++j) {
        const int kb = j * 32;
        if (kb > q) break;
        const float sv = sreg[j];
        for (int i = 0; i < 32; ++i) {
            const int k = kb + i;
            if (k > q) break;
            const float w = __shfl(sv, i, 32);
            const float* vp = Vv + (brow + k) * CD + d0;
            acc0 += w * vp[0];
            acc1 += w * vp[1];
        }
    }
    bf16* yp = Y + (brow + q) * CD + d0;
    yp[0] = (bf16)(acc0 * inv);
    yp[1] = (bf16)(acc1 * inv);
}

// ---------------------------------------------------------------------------
// gather state rows (m = 1 + 3s) of final LN output -> compact [1024,1024]
// ---------------------------------------------------------------------------
__global__ void gather_kernel(const bf16* __restrict__ Hb, bf16* __restrict__ G) {
    const int idx = blockIdx.x * blockDim.x + threadIdx.x;  // 1024*1024 total
    const int gr = idx >> 10;
    const int d  = idx & 1023;
    const int b  = gr >> 8;
    const int s  = gr & 255;
    G[idx] = Hb[((size_t)b * CM + 1 + 3 * s) * CD + d];
}

// ---------------------------------------------------------------------------
extern "C" void kernel_launch(void* const* d_in, const int* in_sizes, int n_in,
                              void* d_out, int out_size, void* d_ws, size_t ws_size,
                              hipStream_t stream) {
    const float* states    = (const float*)d_in[0];
    const int*   actions   = (const int*)  d_in[1];
    const float* rtg       = (const float*)d_in[2];
    const int*   timesteps = (const int*)  d_in[3];
    const float* pos_emb   = (const float*)d_in[4];
    const float* gpos      = (const float*)d_in[5];
    const float* Ws        = (const float*)d_in[6];
    const float* bs        = (const float*)d_in[7];
    const float* Wr        = (const float*)d_in[8];
    const float* br        = (const float*)d_in[9];
    const float* aemb      = (const float*)d_in[10];
    const float* ln1_g     = (const float*)d_in[11];
    const float* ln1_b     = (const float*)d_in[12];
    const float* Wq        = (const float*)d_in[13];
    const float* bq        = (const float*)d_in[14];
    const float* Wk        = (const float*)d_in[15];
    const float* bk        = (const float*)d_in[16];
    const float* Wv        = (const float*)d_in[17];
    const float* bv        = (const float*)d_in[18];
    const float* Wo        = (const float*)d_in[19];
    const float* bo        = (const float*)d_in[20];
    const float* ln2_g     = (const float*)d_in[21];
    const float* ln2_b     = (const float*)d_in[22];
    const float* W1        = (const float*)d_in[23];
    const float* b1        = (const float*)d_in[24];
    const float* W2        = (const float*)d_in[25];
    const float* b2        = (const float*)d_in[26];
    const float* lnf_g     = (const float*)d_in[27];
    const float* lnf_b     = (const float*)d_in[28];
    const float* headW     = (const float*)d_in[29];
    float* out = (float*)d_out;

    // workspace carve-up (deterministic, 256B aligned)
    char* base = (char*)d_ws;
    size_t off = 0;
    auto alloc = [&](size_t bytes) -> char* {
        char* p = base + off;
        off += (bytes + 255) & ~(size_t)255;
        return p;
    };
    float* Xf    = (float*)alloc((size_t)CROWSP * CD * 4);
    bf16*  Hbf   = (bf16*) alloc((size_t)CROWSP * CD * 2);
    float* Qf    = (float*)alloc((size_t)CROWSP * CD * 4);
    float* Kf    = (float*)alloc((size_t)CROWSP * CD * 4);
    float* Vf    = (float*)alloc((size_t)CROWSP * CD * 4);
    bf16*  Ybf   = (bf16*) alloc((size_t)CROWSP * CD * 2);
    bf16*  FFbf  = (bf16*) alloc((size_t)CROWSP * CFF * 2);
    bf16*  Wqb   = (bf16*) alloc((size_t)CD * CD * 2);
    bf16*  Wkb   = (bf16*) alloc((size_t)CD * CD * 2);
    bf16*  Wvb   = (bf16*) alloc((size_t)CD * CD * 2);
    bf16*  Wob   = (bf16*) alloc((size_t)CD * CD * 2);
    bf16*  W1b   = (bf16*) alloc((size_t)CFF * CD * 2);
    bf16*  W2b   = (bf16*) alloc((size_t)CD * CFF * 2);
    bf16*  Gbf   = (bf16*) alloc((size_t)CHROWS * CD * 2);
    bf16*  HWb   = (bf16*) alloc((size_t)CV * CD * 2);

    // init padding rows, embed tokens
    zero_pad_kernel<<<16, 256, 0, stream>>>(Xf, Ybf);
    embed_kernel<<<dim3(CM, CB), 256, 0, stream>>>(states, actions, rtg, timesteps,
                                                   pos_emb, gpos, Ws, bs, Wr, br,
                                                   aemb, Xf);

    const dim3 gemmD  (CD  / 64, CROWSP / 256);   // N=1024 GEMMs on 3072 rows
    const dim3 gemmFF (CFF / 64, CROWSP / 256);   // N=4096
    const dim3 gemmHd (CV  / 64, CHROWS / 256);   // head: N=32000, 1024 rows

    for (int l = 0; l < CL; ++l) {
        // stage this layer's weights as bf16
        f2bf_kernel<<<1024, 256, 0, stream>>>(Wq + (size_t)l * CD * CD,  Wqb, CD * CD);
        f2bf_kernel<<<1024, 256, 0, stream>>>(Wk + (size_t)l * CD * CD,  Wkb, CD * CD);
        f2bf_kernel<<<1024, 256, 0, stream>>>(Wv + (size_t)l * CD * CD,  Wvb, CD * CD);
        f2bf_kernel<<<1024, 256, 0, stream>>>(Wo + (size_t)l * CD * CD,  Wob, CD * CD);
        f2bf_kernel<<<2048, 256, 0, stream>>>(W1 + (size_t)l * CFF * CD, W1b, CFF * CD);
        f2bf_kernel<<<2048, 256, 0, stream>>>(W2 + (size_t)l * CD * CFF, W2b, CD * CFF);

        // LN1 -> bf16
        ln_kernel<<<CROWSP, 256, 0, stream>>>(Xf, ln1_g + l * CD, ln1_b + l * CD, Hbf);
        // Q,K,V projections (fp32 out)
        gemm_bf16_kernel<false><<<gemmD, 256, 0, stream>>>(Hbf, Wqb, bq + l * CD,
                                                           nullptr, Qf, nullptr, CD, CD);
        gemm_bf16_kernel<false><<<gemmD, 256, 0, stream>>>(Hbf, Wkb, bk + l * CD,
                                                           nullptr, Kf, nullptr, CD, CD);
        gemm_bf16_kernel<false><<<gemmD, 256, 0, stream>>>(Hbf, Wvb, bv + l * CD,
                                                           nullptr, Vf, nullptr, CD, CD);
        // causal attention -> bf16
        attn_kernel<<<dim3((CM + 3) / 4, CH, CB), 128, 0, stream>>>(Qf, Kf, Vf, Ybf);
        // output projection + residual -> Xf
        gemm_bf16_kernel<false><<<gemmD, 256, 0, stream>>>(Ybf, Wob, bo + l * CD,
                                                           Xf, Xf, nullptr, CD, CD);
        // LN2 -> bf16
        ln_kernel<<<CROWSP, 256, 0, stream>>>(Xf, ln2_g + l * CD, ln2_b + l * CD, Hbf);
        // FF1 with exact GELU, bf16 out
        gemm_bf16_kernel<true><<<gemmFF, 256, 0, stream>>>(Hbf, W1b, b1 + l * CFF,
                                                           nullptr, nullptr, FFbf, CFF, CD);
        // FF2 + residual -> Xf
        gemm_bf16_kernel<false><<<gemmD, 256, 0, stream>>>(FFbf, W2b, b2 + l * CD,
                                                           Xf, Xf, nullptr, CD, CFF);
    }

    // final LN, gather state rows, LM head (only needed rows -> d_out)
    ln_kernel<<<CROWSP, 256, 0, stream>>>(Xf, lnf_g, lnf_b, Hbf);
    gather_kernel<<<(CHROWS * CD) / 256, 256, 0, stream>>>(Hbf, Gbf);
    f2bf_kernel<<<8192, 256, 0, stream>>>(headW, HWb, CV * CD);
    gemm_bf16_kernel<false><<<gemmHd, 256, 0, stream>>>(Gbf, HWb, nullptr,
                                                        nullptr, out, nullptr, CV, CD);
}
